// QuerySpecificAttentionFixedEmbedClusteringModel_9131100471910
// MI455X (gfx1250) — matrix operand order repaired and checked by reading, not compiled, not run on hardware
//
#include <hip/hip_runtime.h>
#include <math.h>

typedef _Float16 h8   __attribute__((ext_vector_type(8)));
typedef _Float16 v16h __attribute__((ext_vector_type(16)));
typedef float    v8f  __attribute__((ext_vector_type(8)));

#define N_TOK 4096
#define EMBD  768
#define NHEAD 12
#define HD    64
#define KCL   64

// ---------------------------------------------------------------- WMMA helpers

__device__ __forceinline__ v8f wmma16x16x32(v16h a, v16h b, v8f c) {
  return __builtin_amdgcn_wmma_f32_16x16x32_f16(false, a, false, b, (short)0, c,
                                                false, false);
}

// A-fragment (16x32 f16): lane L holds row m=L&15; K = (L>>4)*16 + e
__device__ __forceinline__ v16h load_frag_a(const _Float16* __restrict__ p, int ld) {
  int lane = threadIdx.x & 31;
  const _Float16* q = p + (size_t)(lane & 15) * ld + (lane >> 4) * 16;
  h8 lo = *(const h8*)q;
  h8 hi = *(const h8*)(q + 8);
  return __builtin_shufflevector(lo, hi, 0,1,2,3,4,5,6,7,8,9,10,11,12,13,14,15);
}

// B-fragment (32x16 f16) from a row-major [N][K] source: lane L holds col n=L&15,
// K = (L>>4)*16 + e  (16 contiguous K elements per lane)
__device__ __forceinline__ v16h load_frag_b(const _Float16* __restrict__ p, int ld) {
  int lane = threadIdx.x & 31;
  const _Float16* q = p + (size_t)(lane & 15) * ld + (lane >> 4) * 16;
  h8 lo = *(const h8*)q;
  h8 hi = *(const h8*)(q + 8);
  return __builtin_shufflevector(lo, hi, 0,1,2,3,4,5,6,7,8,9,10,11,12,13,14,15);
}

__device__ __forceinline__ float block_sum(float v, float* red) {
#pragma unroll
  for (int o = 16; o; o >>= 1) v += __shfl_xor(v, o, 32);
  __syncthreads();
  if ((threadIdx.x & 31) == 0) red[threadIdx.x >> 5] = v;
  __syncthreads();
  float t = (threadIdx.x < (blockDim.x >> 5)) ? red[threadIdx.x] : 0.f;
#pragma unroll
  for (int o = 16; o; o >>= 1) t += __shfl_xor(t, o, 32);
  if (threadIdx.x == 0) red[0] = t;
  __syncthreads();
  float r = red[0];
  __syncthreads();
  return r;
}

// ---------------------------------------------------------------- small kernels

__global__ void __launch_bounds__(256) k_f32tof16(const float* __restrict__ s,
                                                  _Float16* __restrict__ d, int n) {
  int i = blockIdx.x * 256 + threadIdx.x;
  if (i < n) d[i] = (_Float16)s[i];
}

// ---------------------------------------------------------------- generic GEMM
// Y[m][o] = act( sum_k A[m][k] * W[o][k] + bias[o] )   (A f16 [M][Kd], W f16 [Nout][Kd])
// Each wave: 32 rows x 64 cols (2 A-frags share 4 B-frags -> 8 WMMA / 6 frag loads).
// Ping-pong double buffering (2x unrolled) -> no register rotation copies.
__global__ void __launch_bounds__(128) k_gemm_nt(const _Float16* __restrict__ A,
                                                 const _Float16* __restrict__ W,
                                                 const float* __restrict__ bias,
                                                 _Float16* __restrict__ Yh,
                                                 float* __restrict__ Yf,
                                                 int Kd, int Nout, int relu) {
  int wave = threadIdx.x >> 5, lane = threadIdx.x & 31;
  int m0 = blockIdx.x * 128 + wave * 32;
  int n0 = blockIdx.y * 64;
  const _Float16* A0 = A + (size_t)m0 * Kd;
  const _Float16* A1 = A + (size_t)(m0 + 16) * Kd;
  const _Float16* W0 = W + (size_t)n0 * Kd;
  v8f acc[2][4] = {};
  v16h a0a, a1a, ba[4], a0b, a1b, bb[4];

  a0a = load_frag_a(A0, Kd);
  a1a = load_frag_a(A1, Kd);
#pragma unroll
  for (int t = 0; t < 4; ++t) ba[t] = load_frag_b(W0 + (size_t)(t * 16) * Kd, Kd);

  int halfSteps = (Kd >> 6);      // Kd/64; Kd is a multiple of 64 here
  int k = 32;
  for (int s = 0; s < halfSteps - 1; ++s, k += 64) {
    a0b = load_frag_a(A0 + k, Kd);
    a1b = load_frag_a(A1 + k, Kd);
#pragma unroll
    for (int t = 0; t < 4; ++t) bb[t] = load_frag_b(W0 + (size_t)(t * 16) * Kd + k, Kd);
#pragma unroll
    for (int t = 0; t < 4; ++t) acc[0][t] = wmma16x16x32(a0a, ba[t], acc[0][t]);
#pragma unroll
    for (int t = 0; t < 4; ++t) acc[1][t] = wmma16x16x32(a1a, ba[t], acc[1][t]);

    a0a = load_frag_a(A0 + k + 32, Kd);
    a1a = load_frag_a(A1 + k + 32, Kd);
#pragma unroll
    for (int t = 0; t < 4; ++t)
      ba[t] = load_frag_b(W0 + (size_t)(t * 16) * Kd + k + 32, Kd);
#pragma unroll
    for (int t = 0; t < 4; ++t) acc[0][t] = wmma16x16x32(a0b, bb[t], acc[0][t]);
#pragma unroll
    for (int t = 0; t < 4; ++t) acc[1][t] = wmma16x16x32(a1b, bb[t], acc[1][t]);
  }
  a0b = load_frag_a(A0 + k, Kd);
  a1b = load_frag_a(A1 + k, Kd);
#pragma unroll
  for (int t = 0; t < 4; ++t) bb[t] = load_frag_b(W0 + (size_t)(t * 16) * Kd + k, Kd);
#pragma unroll
  for (int t = 0; t < 4; ++t) acc[0][t] = wmma16x16x32(a0a, ba[t], acc[0][t]);
#pragma unroll
  for (int t = 0; t < 4; ++t) acc[1][t] = wmma16x16x32(a1a, ba[t], acc[1][t]);
#pragma unroll
  for (int t = 0; t < 4; ++t) acc[0][t] = wmma16x16x32(a0b, bb[t], acc[0][t]);
#pragma unroll
  for (int t = 0; t < 4; ++t) acc[1][t] = wmma16x16x32(a1b, bb[t], acc[1][t]);

  int n = lane & 15, g = lane >> 4;
#pragma unroll
  for (int i = 0; i < 2; ++i) {
#pragma unroll
    for (int t = 0; t < 4; ++t) {
#pragma unroll
      for (int r = 0; r < 8; ++r) {
        int row = m0 + i * 16 + r + 8 * g;
        int col = n0 + t * 16 + n;
        float v = acc[i][t][r] + bias[col];
        if (relu) v = fmaxf(v, 0.f);
        if (Yh) Yh[(size_t)row * Nout + col] = (_Float16)v;
        if (Yf) Yf[(size_t)row * Nout + col] = v;
      }
    }
  }
}

// ---------------------------------------------------------------- flash attention
// qkv f16 [4096][2304]; o f16 [4096][768]. One block = 1 head x 64 query rows.
__global__ void __launch_bounds__(128) k_attn(const _Float16* __restrict__ qkv,
                                              _Float16* __restrict__ o) {
  __shared__ _Float16 VT[64 * 40];     // V tile transposed: [dim][key], padded stride
  __shared__ _Float16 P[4][16 * 32];   // per-wave softmax probs, row-major 16x32
  int h = blockIdx.y, qb = blockIdx.x;
  int wave = threadIdx.x >> 5, lane = threadIdx.x & 31;
  int n = lane & 15, g = lane >> 4;
  int m0 = qb * 64 + wave * 16;
  const _Float16* q_p = qkv + h * HD;
  const _Float16* k_p = qkv + EMBD + h * HD;
  const _Float16* v_p = qkv + 2 * EMBD + h * HD;

  v16h qf0 = load_frag_a(q_p + (size_t)m0 * 2304, 2304);
  v16h qf1 = load_frag_a(q_p + (size_t)m0 * 2304 + 32, 2304);
  v8f accO[4] = {};
  float run_m[8], run_l[8];
#pragma unroll
  for (int r = 0; r < 8; ++r) { run_m[r] = -3.4e38f; run_l[r] = 0.f; }

  for (int kb = 0; kb < N_TOK; kb += 32) {
    __syncthreads();
    { // cooperative transposed stage of V tile [32 keys][64 dims] -> VT[dim][key]
      int key = threadIdx.x & 31, d0 = (threadIdx.x >> 5) * 16;
      const _Float16* src = v_p + (size_t)(kb + key) * 2304 + d0;
      h8 a0 = *(const h8*)src;
      h8 a1 = *(const h8*)(src + 8);
#pragma unroll
      for (int i = 0; i < 8; ++i) {
        VT[(d0 + i) * 40 + key]     = a0[i];
        VT[(d0 + 8 + i) * 40 + key] = a1[i];
      }
    }
    // S = Q (16x64) @ K^T -> two 16x16 f32 tiles (keys kb..kb+31)
    v8f s[2];
#pragma unroll
    for (int t = 0; t < 2; ++t) {
      v8f c = {};
      v16h b0 = load_frag_b(k_p + (size_t)(kb + t * 16) * 2304, 2304);
      c = wmma16x16x32(qf0, b0, c);
      v16h b1 = load_frag_b(k_p + (size_t)(kb + t * 16) * 2304 + 32, 2304);
      c = wmma16x16x32(qf1, b1, c);
      s[t] = c;
    }
    // online softmax per row (rows live in 16-lane halves of the wave32 C layout)
#pragma unroll
    for (int r = 0; r < 8; ++r) {
      float v0 = s[0][r] * 0.125f, v1 = s[1][r] * 0.125f;
      float mx = fmaxf(v0, v1);
#pragma unroll
      for (int off = 1; off < 16; off <<= 1) mx = fmaxf(mx, __shfl_xor(mx, off, 32));
      float nm = fmaxf(run_m[r], mx);
      float p0 = __expf(v0 - nm), p1 = __expf(v1 - nm);
      float ls = p0 + p1;
#pragma unroll
      for (int off = 1; off < 16; off <<= 1) ls += __shfl_xor(ls, off, 32);
      float corr = __expf(run_m[r] - nm);
      run_l[r] = run_l[r] * corr + ls;
      run_m[r] = nm;
#pragma unroll
      for (int fo = 0; fo < 4; ++fo) accO[fo][r] *= corr;
      int prow = r + 8 * g;
      P[wave][prow * 32 + n]      = (_Float16)p0;
      P[wave][prow * 32 + 16 + n] = (_Float16)p1;
    }
    __syncthreads();
    // O += P (16x32) @ V (32x64)
    v16h pf = load_frag_a(&P[wave][0], 32);
#pragma unroll
    for (int fo = 0; fo < 4; ++fo) {
      v16h vf = load_frag_b(VT + fo * 16 * 40, 40);
      accO[fo] = wmma16x16x32(pf, vf, accO[fo]);
    }
  }
#pragma unroll
  for (int fo = 0; fo < 4; ++fo) {
#pragma unroll
    for (int r = 0; r < 8; ++r) {
      int row = m0 + r + 8 * g;
      int col = h * HD + fo * 16 + n;
      o[(size_t)row * EMBD + col] = (_Float16)(accO[fo][r] / run_l[r]);
    }
  }
}

// ---------------------------------------------------------------- residual + LN
// y (in: fc2 output f32, out: transformed X f32, in-place), also emits X f16 and |x|^2
__global__ void __launch_bounds__(256) k_resid_ln(const float* __restrict__ x,
                                                  float* __restrict__ y,
                                                  const float* __restrict__ gam,
                                                  const float* __restrict__ bet,
                                                  _Float16* __restrict__ Xh,
                                                  float* __restrict__ xx) {
  __shared__ float red[8];
  int row = blockIdx.x, tid = threadIdx.x;
  const float* xr = x + (size_t)row * EMBD;
  float* yr = y + (size_t)row * EMBD;
  float z[3]; float s = 0.f;
#pragma unroll
  for (int i = 0; i < 3; ++i) {
    int j = tid + i * 256;
    z[i] = 0.5f * (xr[j] + yr[j]);
    s += z[i];
  }
  float mu = block_sum(s, red) * (1.f / EMBD);
  float vs = 0.f;
#pragma unroll
  for (int i = 0; i < 3; ++i) { float d = z[i] - mu; vs += d * d; }
  float var = block_sum(vs, red) * (1.f / EMBD);
  float inv = rsqrtf(var + 1e-5f);
  float sq = 0.f;
#pragma unroll
  for (int i = 0; i < 3; ++i) {
    int j = tid + i * 256;
    float ov = (z[i] - mu) * inv * gam[j] + bet[j];
    yr[j] = ov;
    Xh[(size_t)row * EMBD + j] = (_Float16)ov;
    sq += ov * ov;
  }
  float tot = block_sum(sq, red);
  if (tid == 0) xx[row] = tot;
}

// ---------------------------------------------------------------- DKM kernels

__global__ void __launch_bounds__(256) k_gather_init(const float* __restrict__ X,
                                                     const int* __restrict__ idx,
                                                     float* __restrict__ Ccur,
                                                     int* __restrict__ done) {
  int c = blockIdx.x;
  int src = idx[c];
  for (int j = threadIdx.x; j < EMBD; j += 256)
    Ccur[(size_t)c * EMBD + j] = X[(size_t)src * EMBD + j];
  if (c == 0 && threadIdx.x == 0) *done = 0;
}

// step prep: publish C (pre-step) to output, f16 convert, row norms, zero accums
__global__ void __launch_bounds__(256) k_step_prep(const int* __restrict__ done,
                                                   const float* __restrict__ Ccur,
                                                   float* __restrict__ outC,
                                                   _Float16* __restrict__ Ch,
                                                   float* __restrict__ cc,
                                                   float* __restrict__ asum,
                                                   float* __restrict__ diffp) {
  if (*done) return;
  __shared__ float red[8];
  int c = blockIdx.x, tid = threadIdx.x;
  float s = 0.f;
  for (int j = tid; j < EMBD; j += 256) {
    float v = Ccur[(size_t)c * EMBD + j];
    outC[(size_t)c * EMBD + j] = v;
    Ch[(size_t)c * EMBD + j] = (_Float16)v;
    s += v * v;
  }
  float tot = block_sum(s, red);
  if (tid == 0) cc[c] = tot;
  if (c == 0 && tid < KCL) asum[tid] = 0.f;
  if (c == 0 && tid == 0) *diffp = 0.f;
}

// assignment: a = softmax(-2*sqrt(max(|x|^2+|c|^2-2 x.c,0)))  via X@C^T WMMA,
// ping-pong double buffered over the 24 k-steps.
__global__ void __launch_bounds__(32) k_assign(const int* __restrict__ done,
                                               const _Float16* __restrict__ Xh,
                                               const _Float16* __restrict__ Ch,
                                               const float* __restrict__ xx,
                                               const float* __restrict__ cc,
                                               float* __restrict__ aout,
                                               _Float16* __restrict__ aT,
                                               float* __restrict__ asum) {
  if (*done) return;
  int lane = threadIdx.x & 31;
  int n = lane & 15, g = lane >> 4;
  int m0 = blockIdx.x * 16;
  const _Float16* Xp = Xh + (size_t)m0 * EMBD;
  v8f acc[4] = {};
  v16h afa, afb, bfa[4], bfb[4];

  afa = load_frag_a(Xp, EMBD);
#pragma unroll
  for (int t = 0; t < 4; ++t) bfa[t] = load_frag_b(Ch + (size_t)(t * 16) * EMBD, EMBD);
  int k = 32;
  for (int s = 0; s < (EMBD >> 6) - 1; ++s, k += 64) {
    afb = load_frag_a(Xp + k, EMBD);
#pragma unroll
    for (int t = 0; t < 4; ++t)
      bfb[t] = load_frag_b(Ch + (size_t)(t * 16) * EMBD + k, EMBD);
#pragma unroll
    for (int t = 0; t < 4; ++t) acc[t] = wmma16x16x32(afa, bfa[t], acc[t]);
    afa = load_frag_a(Xp + k + 32, EMBD);
#pragma unroll
    for (int t = 0; t < 4; ++t)
      bfa[t] = load_frag_b(Ch + (size_t)(t * 16) * EMBD + k + 32, EMBD);
#pragma unroll
    for (int t = 0; t < 4; ++t) acc[t] = wmma16x16x32(afb, bfb[t], acc[t]);
  }
  afb = load_frag_a(Xp + k, EMBD);
#pragma unroll
  for (int t = 0; t < 4; ++t)
    bfb[t] = load_frag_b(Ch + (size_t)(t * 16) * EMBD + k, EMBD);
#pragma unroll
  for (int t = 0; t < 4; ++t) acc[t] = wmma16x16x32(afa, bfa[t], acc[t]);
#pragma unroll
  for (int t = 0; t < 4; ++t) acc[t] = wmma16x16x32(afb, bfb[t], acc[t]);

  float cs[4] = {0.f, 0.f, 0.f, 0.f};
#pragma unroll
  for (int r = 0; r < 8; ++r) {
    int row = m0 + r + 8 * g;
    float xr = xx[row];
    float lg[4]; float mx = -3.4e38f;
#pragma unroll
    for (int t = 0; t < 4; ++t) {
      float sq = xr + cc[t * 16 + n] - 2.f * acc[t][r];
      float d = sqrtf(fmaxf(sq, 0.f));
      lg[t] = -2.f * d;                       // -dist / TEMP, TEMP=0.5
      mx = fmaxf(mx, lg[t]);
    }
#pragma unroll
    for (int off = 1; off < 16; off <<= 1) mx = fmaxf(mx, __shfl_xor(mx, off, 32));
    float se = 0.f;
#pragma unroll
    for (int t = 0; t < 4; ++t) { lg[t] = __expf(lg[t] - mx); se += lg[t]; }
#pragma unroll
    for (int off = 1; off < 16; off <<= 1) se += __shfl_xor(se, off, 32);
    float inv = 1.f / se;
#pragma unroll
    for (int t = 0; t < 4; ++t) {
      float av = lg[t] * inv;
      aout[(size_t)row * KCL + t * 16 + n] = av;
      aT[(size_t)(t * 16 + n) * N_TOK + row] = (_Float16)av;
      cs[t] += av;
    }
  }
#pragma unroll
  for (int t = 0; t < 4; ++t) {
    cs[t] += __shfl_xor(cs[t], 16, 32);
    if (g == 0) atomicAdd(&asum[t * 16 + n], cs[t]);
  }
}

// update: Cacc = a^T @ X  (64 x 768, K=4096), ping-pong pipelined strided B gather
__global__ void __launch_bounds__(32) k_update(const int* __restrict__ done,
                                               const _Float16* __restrict__ aT,
                                               const _Float16* __restrict__ Xh,
                                               float* __restrict__ Cacc) {
  if (*done) return;
  int lane = threadIdx.x & 31;
  int n = lane & 15, g = lane >> 4;
  int m0 = blockIdx.y * 16;   // cluster tile
  int n0 = blockIdx.x * 16;   // feature tile
  const _Float16* aTp = aT + (size_t)m0 * N_TOK;
  const _Float16* Xcol = Xh + n0 + n;
  v8f acc = {};
  v16h afa, afb, bfa, bfb;

  afa = load_frag_a(aTp, N_TOK);
#pragma unroll
  for (int e = 0; e < 16; ++e) bfa[e] = Xcol[(size_t)(g * 16 + e) * EMBD];
  int k = 32;
  for (int s = 0; s < (N_TOK >> 6) - 1; ++s, k += 64) {
    afb = load_frag_a(aTp + k, N_TOK);
#pragma unroll
    for (int e = 0; e < 16; ++e) bfb[e] = Xcol[(size_t)(k + g * 16 + e) * EMBD];
    acc = wmma16x16x32(afa, bfa, acc);
    afa = load_frag_a(aTp + k + 32, N_TOK);
#pragma unroll
    for (int e = 0; e < 16; ++e) bfa[e] = Xcol[(size_t)(k + 32 + g * 16 + e) * EMBD];
    acc = wmma16x16x32(afb, bfb, acc);
  }
  afb = load_frag_a(aTp + k, N_TOK);
#pragma unroll
  for (int e = 0; e < 16; ++e) bfb[e] = Xcol[(size_t)(k + g * 16 + e) * EMBD];
  acc = wmma16x16x32(afa, bfa, acc);
  acc = wmma16x16x32(afb, bfb, acc);
#pragma unroll
  for (int r = 0; r < 8; ++r)
    Cacc[(size_t)(m0 + r + 8 * g) * EMBD + n0 + n] = acc[r];
}

__global__ void __launch_bounds__(256) k_finalize(const int* __restrict__ done,
                                                  const float* __restrict__ Cacc,
                                                  const float* __restrict__ asum,
                                                  float* __restrict__ Ccur,
                                                  float* __restrict__ diffp) {
  if (*done) return;
  __shared__ float red[8];
  int c = blockIdx.x;
  float denom = 1.f / (asum[c] + 1e-6f);
  float dl = 0.f;
  for (int j = threadIdx.x; j < EMBD; j += 256) {
    float v = Cacc[(size_t)c * EMBD + j] * denom;
    dl += fabsf(v - Ccur[(size_t)c * EMBD + j]);
    Ccur[(size_t)c * EMBD + j] = v;
  }
  float tot = block_sum(dl, red);
  if (threadIdx.x == 0) atomicAdd(diffp, tot);
}

__global__ void k_check(int* done, const float* diffp) {
  if (!*done && *diffp <= 1e-4f) *done = 1;
}

// ---------------------------------------------------------------- host launcher

extern "C" void kernel_launch(void* const* d_in, const int* in_sizes, int n_in,
                              void* d_out, int out_size, void* d_ws, size_t ws_size,
                              hipStream_t stream) {
  (void)in_sizes; (void)n_in; (void)out_size; (void)ws_size;
  const float* emb   = (const float*)d_in[0];
  const float* fc1_w = (const float*)d_in[1];
  const float* fc1_b = (const float*)d_in[2];
  const float* fc2_w = (const float*)d_in[3];
  const float* fc2_b = (const float*)d_in[4];
  const float* inp_w = (const float*)d_in[5];
  const float* inp_b = (const float*)d_in[6];
  const float* out_w = (const float*)d_in[7];
  const float* out_b = (const float*)d_in[8];
  const float* ln_g  = (const float*)d_in[9];
  const float* ln_b  = (const float*)d_in[10];
  const int*   idx   = (const int*)d_in[11];

  char* ws = (char*)d_ws;
  const size_t OFF_CTL  = 0;                          // done(int) + diff(float)
  const size_t OFF_XX   = 256;                        // 4096 f32
  const size_t OFF_CC   = OFF_XX   + 16384;           // 64 f32
  const size_t OFF_ASUM = OFF_CC   + 256;             // 64 f32
  const size_t OFF_CCUR = OFF_ASUM + 256;             // 64x768 f32
  const size_t OFF_CH   = OFF_CCUR + 196608;          // 64x768 f16
  const size_t OFF_CACC = OFF_CH   + 98304;           // 64x768 f32
  const size_t OFF_AT   = OFF_CACC + 196608;          // 64x4096 f16
  const size_t OFF_EMBH = OFF_AT   + 524288;          // 4096x768 f16 (emb, later o2)
  const size_t OFF_W1H  = OFF_EMBH + 6291456;
  const size_t OFF_WPH  = OFF_W1H  + 1179648;
  const size_t OFF_WOH  = OFF_WPH  + 3538944;
  const size_t OFF_W2H  = OFF_WOH  + 1179648;
  const size_t OFF_H    = OFF_W2H  + 1179648;         // h f16, later o f16
  const size_t OFF_QKV  = OFF_H    + 6291456;         // 4096x2304 f16
  const size_t OFF_Y    = OFF_QKV  + 18874368;        // y f32, later X f32
  const size_t OFF_XH   = OFF_Y    + 12582912;        // X f16

  int*      done  = (int*)(ws + OFF_CTL);
  float*    diffp = (float*)(ws + OFF_CTL + 4);
  float*    xx    = (float*)(ws + OFF_XX);
  float*    cc    = (float*)(ws + OFF_CC);
  float*    asum  = (float*)(ws + OFF_ASUM);
  float*    Ccur  = (float*)(ws + OFF_CCUR);
  _Float16* Ch    = (_Float16*)(ws + OFF_CH);
  float*    Cacc  = (float*)(ws + OFF_CACC);
  _Float16* aTh   = (_Float16*)(ws + OFF_AT);
  _Float16* embh  = (_Float16*)(ws + OFF_EMBH);
  _Float16* w1h   = (_Float16*)(ws + OFF_W1H);
  _Float16* wph   = (_Float16*)(ws + OFF_WPH);
  _Float16* woh   = (_Float16*)(ws + OFF_WOH);
  _Float16* w2h   = (_Float16*)(ws + OFF_W2H);
  _Float16* hbuf  = (_Float16*)(ws + OFF_H);
  _Float16* qkvh  = (_Float16*)(ws + OFF_QKV);
  float*    ybuf  = (float*)(ws + OFF_Y);
  _Float16* Xh    = (_Float16*)(ws + OFF_XH);

  // ---- f32 -> f16 conversions
  k_f32tof16<<<(N_TOK*EMBD+255)/256, 256, 0, stream>>>(emb,   embh, N_TOK*EMBD);
  k_f32tof16<<<(EMBD*EMBD +255)/256, 256, 0, stream>>>(fc1_w, w1h,  EMBD*EMBD);
  k_f32tof16<<<(3*EMBD*EMBD+255)/256,256, 0, stream>>>(inp_w, wph,  3*EMBD*EMBD);
  k_f32tof16<<<(EMBD*EMBD +255)/256, 256, 0, stream>>>(out_w, woh,  EMBD*EMBD);
  k_f32tof16<<<(EMBD*EMBD +255)/256, 256, 0, stream>>>(fc2_w, w2h,  EMBD*EMBD);

  // ---- transformer block (GEMM blocks cover 128 rows x 64 cols)
  k_gemm_nt<<<dim3(32,12), 128, 0, stream>>>(embh, w1h, fc1_b, hbuf, nullptr, EMBD, EMBD, 1);
  k_gemm_nt<<<dim3(32,36), 128, 0, stream>>>(hbuf, wph, inp_b, qkvh, nullptr, EMBD, 3*EMBD, 0);
  k_attn   <<<dim3(64,12), 128, 0, stream>>>(qkvh, hbuf);                       // o -> hbuf
  k_gemm_nt<<<dim3(32,12), 128, 0, stream>>>(hbuf, woh, out_b, embh, nullptr, EMBD, EMBD, 0); // o2
  k_gemm_nt<<<dim3(32,12), 128, 0, stream>>>(embh, w2h, fc2_b, nullptr, ybuf, EMBD, EMBD, 1); // y
  k_resid_ln<<<N_TOK, 256, 0, stream>>>(emb, ybuf, ln_g, ln_b, Xh, xx);         // X in ybuf

  // ---- DKM clustering: fixed 1 + MAX_ITER steps, device-side convergence flag
  k_gather_init<<<KCL, 256, 0, stream>>>(ybuf, idx, Ccur, done);
  float* outC = (float*)d_out;
  float* outA = outC + (size_t)KCL * EMBD;
  for (int it = 0; it < 101; ++it) {
    k_step_prep<<<KCL, 256, 0, stream>>>(done, Ccur, outC, Ch, cc, asum, diffp);
    k_assign<<<N_TOK/16, 32, 0, stream>>>(done, Xh, Ch, xx, cc, outA, aTh, asum);
    k_update<<<dim3(EMBD/16, KCL/16), 32, 0, stream>>>(done, aTh, Xh, Cacc);
    k_finalize<<<KCL, 256, 0, stream>>>(done, Cacc, asum, Ccur, diffp);
    k_check<<<1, 1, 0, stream>>>(done, diffp);
  }
}
